// CausalSelfAttnBlock_23081154249030
// MI455X (gfx1250) — matrix operand order repaired and checked by reading, not compiled, not run on hardware
//
#include <hip/hip_runtime.h>
#include <hip/hip_bf16.h>

// B=2, T=2048, D=1024, H=16, HD=64
#define TB 2
#define TT 2048
#define TD 1024
#define TH 16
#define THD 64

typedef unsigned short u16;
typedef unsigned int   u32;
typedef __bf16 bf16_t;
typedef bf16_t v16bf __attribute__((ext_vector_type(16)));
typedef float  v8f   __attribute__((ext_vector_type(8)));

union Frag {
    v16bf bf;
    u32   d[8];
    uint4 q[2];
    u16   h[16];
};

__device__ __forceinline__ v8f v8f_zero() {
    v8f z = {0.f, 0.f, 0.f, 0.f, 0.f, 0.f, 0.f, 0.f};
    return z;
}

__device__ __forceinline__ u16 f2bf(float x) {
    u32 u = __float_as_uint(x);
    return (u16)((u + 0x7FFFu + ((u >> 16) & 1u)) >> 16); // RNE
}

// CDNA5 async global->LDS copy (16B per lane), tracked by ASYNCcnt.
__device__ __forceinline__ void async_ld_b128(u16* lds, const u16* g) {
    u32 l = (u32)(size_t)lds;   // low 32 bits of generic ptr == LDS byte offset
    asm volatile("global_load_async_to_lds_b128 %0, %1, off"
                 :: "v"(l), "v"(g) : "memory");
}
__device__ __forceinline__ void wait_async0() {
    asm volatile("s_wait_asynccnt 0x0" ::: "memory");
}

// A fragment (16x32 bf16, MxK). Rows at base (row stride ld, in u16 units).
// CDNA5 7.12.2: lane L holds row M=L%16; VGPR v holds K pair
// ((v<4)?0:16) + (v&3)*2 + (L/16)*8.  (pairs coalesce into 2x ds_load_b128)
__device__ __forceinline__ Frag load_a_frag(const u16* base, int ld, int lane) {
    Frag f;
    const u16* rp = base + (lane & 15) * ld;
    const int half8 = (lane >> 4) * 8;
#pragma unroll
    for (int v = 0; v < 8; ++v) {
        int k = ((v < 4) ? 0 : 16) + (v & 3) * 2 + half8;
        f.d[v] = *(const u32*)(rp + k);
    }
    return f;
}

// B fragment (32x16 bf16, KxN) from N-major memory: element (k,n) at
// base[n*ld + k]. Lane L: column N=L%16, K=(L/16)*8 + j (j<8), 16 + (L/16)*8 + (j-8) (j>=8).
__device__ __forceinline__ Frag load_b_frag(const u16* base, int ld, int lane) {
    Frag f;
    const u16* p = base + (size_t)(lane & 15) * ld + (lane >> 4) * 8;
    f.q[0] = *(const uint4*)(p);
    f.q[1] = *(const uint4*)(p + 16);
    return f;
}

__device__ __forceinline__ v8f wmma_bf16(const Frag& a, const Frag& b, v8f c) {
    return __builtin_amdgcn_wmma_f32_16x16x32_bf16(
        false, a.bf, false, b.bf, (short)0, c, false, false);
}

// ---------------------------------------------------------------------------
// fp32 -> bf16 conversion (4 elems/thread)
// ---------------------------------------------------------------------------
__global__ void __launch_bounds__(256)
cvt_bf16_kernel(const float* __restrict__ in, u16* __restrict__ out, int n4) {
    int i = (blockIdx.x * 256 + threadIdx.x);
    if (i < n4) {
        float4 v = *(const float4*)(in + i * 4);
        uint2 o;
        o.x = (u32)f2bf(v.x) | ((u32)f2bf(v.y) << 16);
        o.y = (u32)f2bf(v.z) | ((u32)f2bf(v.w) << 16);
        *(uint2*)(out + i * 4) = o;
    }
}

// ===========================================================================
// Double-buffered 64x128 WMMA GEMM building blocks (128 thr = 4 waves,
// each wave 32x64 = 2x4 WMMA tiles, K-step 64).
//   A tile : async global->LDS (ASYNCcnt), row-major As[m][k]
//   B tile : global->VGPR during previous iteration's compute, packed into
//            transposed Bs[n][k] at the top of the consuming iteration.
// ===========================================================================
struct BRegs { u16 r0[32]; u16 r1[32]; };   // two k-rows x 32 n

__device__ __forceinline__ BRegs load_b_regs(const u16* W, int ldw,
                                             int kk, int n0, int tid) {
    // pair = tid>>2 selects k-pair, (tid&3)*32 selects 32-wide n segment
    const int kp = (tid >> 2) * 2;
    const int nseg = (tid & 3) * 32;
    const u16* s0 = W + (size_t)(kk + kp) * ldw + n0 + nseg;
    const u16* s1 = s0 + ldw;
    BRegs b;
#pragma unroll
    for (int i = 0; i < 4; ++i) {
        *(uint4*)&b.r0[i * 8] = *(const uint4*)(s0 + i * 8);
        *(uint4*)&b.r1[i * 8] = *(const uint4*)(s1 + i * 8);
    }
    return b;
}

__device__ __forceinline__ void pack_store_b(u16* Bs, const BRegs& b, int tid) {
    const int kp = (tid >> 2) * 2;
    const int nseg = (tid & 3) * 32;
#pragma unroll
    for (int i = 0; i < 32; ++i) {
        u32 pk = (u32)b.r0[i] | ((u32)b.r1[i] << 16);
        *(u32*)&Bs[(size_t)(nseg + i) * 64 + kp] = pk;
    }
}

__device__ __forceinline__ void async_stage_a(u16* As, const u16* X, int lda,
                                              int m0, int kk, int tid) {
    const int row = tid >> 1, colh = (tid & 1) * 32;
    const u16* src = X + (size_t)(m0 + row) * lda + kk + colh;
    u16* dst = As + row * 64 + colh;
#pragma unroll
    for (int i = 0; i < 4; ++i) async_ld_b128(dst + i * 8, src + i * 8);
}

// ---------------------------------------------------------------------------
// QKV projection: [4096,1024] x [1024,3072] + bias.
// Q,K stored [B,H,T,HD]; V stored TRANSPOSED [B,H,HD,T] so the attention
// P@V B-fragments are contiguous global loads.
// ---------------------------------------------------------------------------
__global__ void __launch_bounds__(128)
qkv_gemm_kernel(const u16* __restrict__ X, const u16* __restrict__ W,
                const float* __restrict__ bias,
                u16* __restrict__ Q, u16* __restrict__ K, u16* __restrict__ V) {
    __shared__ u16 As[2][64 * 64];        // [m][k]
    __shared__ u16 Bs[2][128 * 64];       // [n][k]
    const int tid = threadIdx.x;
    const int lane = tid & 31, wave = tid >> 5;
    const int wm = (wave >> 1) * 32, wn = (wave & 1) * 64;
    const int m0 = blockIdx.x * 64;
    const int n0 = blockIdx.y * 128;

    v8f acc[2][4];
#pragma unroll
    for (int i = 0; i < 2; ++i)
#pragma unroll
        for (int j = 0; j < 4; ++j) acc[i][j] = v8f_zero();

    BRegs br = load_b_regs(W, 3 * TD, 0, n0, tid);
    async_stage_a(As[0], X, TD, m0, 0, tid);

    for (int it = 0; it < 16; ++it) {
        const int p = it & 1;
        pack_store_b(Bs[p], br, tid);
        wait_async0();
        __syncthreads();                   // buf p fully staged, all waves
        if (it < 15) {                     // overlap next tile with compute
            br = load_b_regs(W, 3 * TD, (it + 1) * 64, n0, tid);
            async_stage_a(As[1 - p], X, TD, m0, (it + 1) * 64, tid);
        }
#pragma unroll
        for (int kf = 0; kf < 2; ++kf) {
            Frag a0 = load_a_frag(&As[p][(wm + 0) * 64 + kf * 32], 64, lane);
            Frag a1 = load_a_frag(&As[p][(wm + 16) * 64 + kf * 32], 64, lane);
#pragma unroll
            for (int sn = 0; sn < 4; ++sn) {
                Frag b = load_b_frag(&Bs[p][(size_t)(wn + sn * 16) * 64 + kf * 32],
                                     64, lane);
                acc[0][sn] = wmma_bf16(a0, b, acc[0][sn]);
                acc[1][sn] = wmma_bf16(a1, b, acc[1][sn]);
            }
        }
        __syncthreads();                   // all reads of buf p done
    }

    const int mrow = (lane >> 4) * 8;
    const int ncol = lane & 15;
#pragma unroll
    for (int sm = 0; sm < 2; ++sm)
#pragma unroll
        for (int sn = 0; sn < 4; ++sn)
#pragma unroll
            for (int r = 0; r < 8; ++r) {
                int m = m0 + wm + sm * 16 + mrow + r;
                int n = n0 + wn + sn * 16 + ncol;
                float v = acc[sm][sn][r] + bias[n];
                int g = n >> 10;          // 0=q 1=k 2=v
                int d = n & 1023;
                int hh = d >> 6, hd = d & 63;
                int bb = m >> 11, tt = m & 2047;
                u16 val = f2bf(v);
                if (g == 0) {
                    Q[(((bb << 4) + hh) * TT + tt) * THD + hd] = val;
                } else if (g == 1) {
                    K[(((bb << 4) + hh) * TT + tt) * THD + hd] = val;
                } else { // V transposed: [B,H,HD,T]
                    V[(((bb << 4) + hh) * THD + hd) * TT + tt] = val;
                }
            }
}

// ---------------------------------------------------------------------------
// Streaming causal attention. One wave per 32-query tile, 32-key chunks.
// Matches reference exactly: score=(qk+mask)/8, no max-subtract,
// weights = exp(score)/(sum+1e-9); causal mask applied analytically.
// Work-balanced tile mapping: waves g and g^1 get tiles u and 2047-u so each
// 4-wave block has constant total chunk count.
// ---------------------------------------------------------------------------
__global__ void __launch_bounds__(128)
attn_kernel(const u16* __restrict__ Q, const u16* __restrict__ K,
            const u16* __restrict__ V, u16* __restrict__ Y) {
    __shared__ u16 Pbuf[4][32 * 32];     // per-wave P (32q x 32k)
    const int lane = threadIdx.x & 31;
    const int w = threadIdx.x >> 5;
    const int g = blockIdx.x * 4 + w;    // 0..2047
    const int u = g >> 1;
    const int lin = (g & 1) ? (2047 - u) : u;
    const int qt = lin & 63;             // T/32 = 64 q-tiles per (b,h)
    const int bh = lin >> 6;             // 0..31
    const int qbase = qt * 32;
    const u16* qp = Q + (size_t)(bh * TT + qbase) * THD;
    const u16* kp = K + (size_t)bh * TT * THD;
    const u16* vp = V + (size_t)bh * THD * TT;   // transposed [HD][T]

    // Q fragments: 2 m-tiles x 2 feature halves, held across the key loop
    Frag qa[2][2];
#pragma unroll
    for (int sm = 0; sm < 2; ++sm)
#pragma unroll
        for (int fh = 0; fh < 2; ++fh)
            qa[sm][fh] = load_a_frag(qp + sm * 16 * THD + fh * 32, THD, lane);

    v8f oacc[2][4];
#pragma unroll
    for (int sm = 0; sm < 2; ++sm)
#pragma unroll
        for (int t4 = 0; t4 < 4; ++t4) oacc[sm][t4] = v8f_zero();
    float dp[2][8];
#pragma unroll
    for (int sm = 0; sm < 2; ++sm)
#pragma unroll
        for (int r = 0; r < 8; ++r) dp[sm][r] = 0.f;

    const int half8 = (lane >> 4) * 8;
    const int ncol = lane & 15;
    const int nchunk = (qbase >> 5) + 1;       // keys 0..qbase+31 (causal)

    for (int c = 0; c < nchunk; ++c) {
        const int kbase = c * 32;
        // K^T fragments: [n-tile][feature-half], reused by both m-tiles
        Frag kb[2][2];
#pragma unroll
        for (int nt = 0; nt < 2; ++nt)
#pragma unroll
            for (int fh = 0; fh < 2; ++fh)
                kb[nt][fh] = load_b_frag(
                    kp + (size_t)(kbase + nt * 16) * THD + fh * 32, THD, lane);
        // V fragments (32 keys x 16 feats each), reused by both m-tiles
        Frag vb[4];
#pragma unroll
        for (int t4 = 0; t4 < 4; ++t4)
            vb[t4] = load_b_frag(vp + (size_t)(t4 * 16) * TT + kbase, TT, lane);

        // scores
        v8f s[2][2];
#pragma unroll
        for (int sm = 0; sm < 2; ++sm)
#pragma unroll
            for (int nt = 0; nt < 2; ++nt) {
                v8f a = v8f_zero();
                a = wmma_bf16(qa[sm][0], kb[nt][0], a);
                a = wmma_bf16(qa[sm][1], kb[nt][1], a);
                s[sm][nt] = a;
            }

        // P = exp(s/8) with causal mask; per-lane partial row sums
#pragma unroll
        for (int sm = 0; sm < 2; ++sm)
#pragma unroll
            for (int r = 0; r < 8; ++r) {
                int m = qbase + sm * 16 + half8 + r;
#pragma unroll
                for (int nt = 0; nt < 2; ++nt) {
                    int n = kbase + nt * 16 + ncol;
                    float p = (n <= m) ? __expf(s[sm][nt][r] * 0.125f) : 0.0f;
                    dp[sm][r] += p;
                    Pbuf[w][(sm * 16 + half8 + r) * 32 + nt * 16 + ncol] = f2bf(p);
                }
            }
        asm volatile("" ::: "memory");   // keep LDS store->load order (in-order DS)

#pragma unroll
        for (int sm = 0; sm < 2; ++sm) {
            Frag pa = load_a_frag(&Pbuf[w][sm * 16 * 32], 32, lane);
#pragma unroll
            for (int t4 = 0; t4 < 4; ++t4)
                oacc[sm][t4] = wmma_bf16(pa, vb[t4], oacc[sm][t4]);
        }
    }

    // reduce row sums across the 16 lanes of each half-wave
#pragma unroll
    for (int sm = 0; sm < 2; ++sm)
#pragma unroll
        for (int r = 0; r < 8; ++r) {
            float d = dp[sm][r];
            d += __shfl_xor(d, 1, 32);
            d += __shfl_xor(d, 2, 32);
            d += __shfl_xor(d, 4, 32);
            d += __shfl_xor(d, 8, 32);
            dp[sm][r] = d;
        }

    const int bb = bh >> 4, hh = bh & 15;
#pragma unroll
    for (int sm = 0; sm < 2; ++sm)
#pragma unroll
        for (int t4 = 0; t4 < 4; ++t4)
#pragma unroll
            for (int r = 0; r < 8; ++r) {
                int m = qbase + sm * 16 + half8 + r;
                int feat = t4 * 16 + ncol;
                float val = oacc[sm][t4][r] / (dp[sm][r] + 1e-9f);
                Y[(size_t)(bb * TT + m) * TD + hh * THD + feat] = f2bf(val);
            }
}

// ---------------------------------------------------------------------------
// Output projection + residual: Z = x + Y @ W_out + b_out (fp32)
// ---------------------------------------------------------------------------
__global__ void __launch_bounds__(128)
outproj_gemm_kernel(const u16* __restrict__ Yb, const u16* __restrict__ W,
                    const float* __restrict__ bias, const float* __restrict__ Xf,
                    float* __restrict__ Z) {
    __shared__ u16 As[2][64 * 64];
    __shared__ u16 Bs[2][128 * 64];
    const int tid = threadIdx.x;
    const int lane = tid & 31, wave = tid >> 5;
    const int wm = (wave >> 1) * 32, wn = (wave & 1) * 64;
    const int m0 = blockIdx.x * 64;
    const int n0 = blockIdx.y * 128;

    v8f acc[2][4];
#pragma unroll
    for (int i = 0; i < 2; ++i)
#pragma unroll
        for (int j = 0; j < 4; ++j) acc[i][j] = v8f_zero();

    BRegs br = load_b_regs(W, TD, 0, n0, tid);
    async_stage_a(As[0], Yb, TD, m0, 0, tid);

    for (int it = 0; it < 16; ++it) {
        const int p = it & 1;
        pack_store_b(Bs[p], br, tid);
        wait_async0();
        __syncthreads();
        if (it < 15) {
            br = load_b_regs(W, TD, (it + 1) * 64, n0, tid);
            async_stage_a(As[1 - p], Yb, TD, m0, (it + 1) * 64, tid);
        }
#pragma unroll
        for (int kf = 0; kf < 2; ++kf) {
            Frag a0 = load_a_frag(&As[p][(wm + 0) * 64 + kf * 32], 64, lane);
            Frag a1 = load_a_frag(&As[p][(wm + 16) * 64 + kf * 32], 64, lane);
#pragma unroll
            for (int sn = 0; sn < 4; ++sn) {
                Frag b = load_b_frag(&Bs[p][(size_t)(wn + sn * 16) * 64 + kf * 32],
                                     64, lane);
                acc[0][sn] = wmma_bf16(a0, b, acc[0][sn]);
                acc[1][sn] = wmma_bf16(a1, b, acc[1][sn]);
            }
        }
        __syncthreads();
    }

    const int mrow = (lane >> 4) * 8;
    const int ncol = lane & 15;
#pragma unroll
    for (int sm = 0; sm < 2; ++sm)
#pragma unroll
        for (int sn = 0; sn < 4; ++sn)
#pragma unroll
            for (int r = 0; r < 8; ++r) {
                int m = m0 + wm + sm * 16 + mrow + r;
                int n = n0 + wn + sn * 16 + ncol;
                size_t idx = (size_t)m * TD + n;
                Z[idx] = acc[sm][sn][r] + Xf[idx] + bias[n];
            }
}

// ---------------------------------------------------------------------------
// LayerNorm over last dim (1024) per row
// ---------------------------------------------------------------------------
__global__ void __launch_bounds__(256)
ln_kernel(const float* __restrict__ Z, const float* __restrict__ gamma,
          const float* __restrict__ beta, float* __restrict__ out) {
    __shared__ float rs[256], rs2[256];
    const int row = blockIdx.x;
    const float* zr = Z + (size_t)row * TD;
    float s = 0.f, s2 = 0.f;
    for (int i = threadIdx.x; i < TD; i += 256) {
        float v = zr[i];
        s += v;
        s2 += v * v;
    }
    rs[threadIdx.x] = s;
    rs2[threadIdx.x] = s2;
    __syncthreads();
    for (int st = 128; st > 0; st >>= 1) {
        if (threadIdx.x < st) {
            rs[threadIdx.x] += rs[threadIdx.x + st];
            rs2[threadIdx.x] += rs2[threadIdx.x + st];
        }
        __syncthreads();
    }
    float mu = rs[0] * (1.0f / TD);
    float var = rs2[0] * (1.0f / TD) - mu * mu;
    float inv = rsqrtf(var + 1e-5f);
    for (int i = threadIdx.x; i < TD; i += 256)
        out[(size_t)row * TD + i] = (zr[i] - mu) * inv * gamma[i] + beta[i];
}

// ---------------------------------------------------------------------------
extern "C" void kernel_launch(void* const* d_in, const int* in_sizes, int n_in,
                              void* d_out, int out_size, void* d_ws, size_t ws_size,
                              hipStream_t stream) {
    (void)in_sizes; (void)n_in; (void)out_size; (void)ws_size;
    const float* x      = (const float*)d_in[0];
    // d_in[1] = attn_mask: causal, applied analytically inside attn_kernel
    const float* W_proj = (const float*)d_in[2];
    const float* b_proj = (const float*)d_in[3];
    const float* W_out  = (const float*)d_in[4];
    const float* b_out  = (const float*)d_in[5];
    const float* gamma  = (const float*)d_in[6];
    const float* beta   = (const float*)d_in[7];
    float* out = (float*)d_out;

    const size_t N_X  = (size_t)TB * TT * TD;       // 4,194,304
    const size_t N_WP = (size_t)TD * 3 * TD;        // 3,145,728
    const size_t N_WO = (size_t)TD * TD;            // 1,048,576

    char* ws = (char*)d_ws;
    size_t off = 0;
    u16* xbf  = (u16*)(ws + off); off += N_X * 2;
    u16* wpbf = (u16*)(ws + off); off += N_WP * 2;
    u16* wobf = (u16*)(ws + off); off += N_WO * 2;
    u16* qbf  = (u16*)(ws + off); off += N_X * 2;
    u16* kbf  = (u16*)(ws + off); off += N_X * 2;
    u16* vbf  = (u16*)(ws + off); off += N_X * 2;   // transposed [B,H,HD,T]
    u16* ybf  = (u16*)(ws + off); off += N_X * 2;
    float* z  = (float*)(ws + off); off += N_X * 4; // total 64 MiB

    cvt_bf16_kernel<<<(int)(N_X / 4 / 256), 256, 0, stream>>>(x, xbf, (int)(N_X / 4));
    cvt_bf16_kernel<<<(int)(N_WP / 4 / 256), 256, 0, stream>>>(W_proj, wpbf, (int)(N_WP / 4));
    cvt_bf16_kernel<<<(int)(N_WO / 4 / 256), 256, 0, stream>>>(W_out, wobf, (int)(N_WO / 4));

    qkv_gemm_kernel<<<dim3(64, 24), 128, 0, stream>>>(xbf, wpbf, b_proj, qbf, kbf, vbf);

    attn_kernel<<<512, 128, 0, stream>>>(qbf, kbf, vbf, ybf);

    outproj_gemm_kernel<<<dim3(64, 8), 128, 0, stream>>>(ybf, wobf, b_out, x, z);

    ln_kernel<<<TB * TT, 256, 0, stream>>>(z, gamma, beta, out);
}
